// DurationPredictor_1675037246176
// MI455X (gfx1250) — matrix operand order repaired and checked by reading, not compiled, not run on hardware
//
#include <hip/hip_runtime.h>
#include <stdint.h>

typedef __attribute__((ext_vector_type(16))) __bf16 v16bf;
typedef __attribute__((ext_vector_type(8)))  __bf16 v8bf;
typedef __attribute__((ext_vector_type(8)))  float  v8f;

static constexpr int B_ = 32, T_ = 1024, C_ = 512;
static constexpr int TPAD   = T_ + 2;      // zero row before and after each sequence
static constexpr int KSTEPS = 48;          // 1536 / 32
static constexpr int LMAX   = 4096;        // T * MAX_DUR
static constexpr size_t XPAD_ELEMS = (size_t)B_ * TPAD * C_;        // bf16 elems
static constexpr size_t BWS_ELEMS  = (size_t)KSTEPS * 32 * 512;     // 786432 bf16 elems
static constexpr size_t Y_ELEMS    = (size_t)B_ * T_ * C_;          // f32 elems

__device__ __forceinline__ __bf16 f2bf(float f) { return (__bf16)f; }

// ---------- fp32 batch -> zero-padded bf16 [B][T+2][C] ----------
__global__ void pad_convert_kernel(const float* __restrict__ batch, __bf16* __restrict__ xpad)
{
  size_t e4 = ((size_t)blockIdx.x * blockDim.x + threadIdx.x) * 4;
  if (e4 >= XPAD_ELEMS) return;
  size_t b   = e4 / ((size_t)TPAD * C_);
  size_t rem = e4 - b * (size_t)TPAD * C_;
  int r = (int)(rem / C_);
  int c = (int)(rem % C_);
  __bf16* dst = xpad + e4;
  if (r == 0 || r == TPAD - 1) {
    dst[0] = f2bf(0.f); dst[1] = f2bf(0.f); dst[2] = f2bf(0.f); dst[3] = f2bf(0.f);
  } else {
    const float4 v = *(const float4*)(batch + ((size_t)b * T_ + (r - 1)) * C_ + c);
    dst[0] = f2bf(v.x); dst[1] = f2bf(v.y); dst[2] = f2bf(v.z); dst[3] = f2bf(v.w);
  }
}

// zero the two pad rows of the second padded buffer (body rows written by LN1)
__global__ void zero_pad_kernel(__bf16* __restrict__ xpad)
{
  int idx = blockIdx.x * blockDim.x + threadIdx.x;   // B*2*C = 32768
  int b  = idx >> 10;
  int rr = idx & 1023;
  int r  = (rr < C_) ? 0 : (TPAD - 1);
  int c  = rr & 511;
  xpad[((size_t)b * TPAD + r) * C_ + c] = f2bf(0.f);
}

// ---------- conv weights (Cout,Cin,3) -> WMMA B-fragment-ordered bf16 ----------
// Layout: [(ks*32 + ntile)*512 + lane*16 + elem]; lane's 16 bf16 are
// Bmat[K = ks*32 + (lane>>4)*16 + elem][N = ntile*16 + (lane&15)],
// with Kd index kd = k*C + ci (A row is 3 contiguous padded time rows).
__global__ void weight_prep_kernel(const float* __restrict__ w, __bf16* __restrict__ bws)
{
  int idx = blockIdx.x * blockDim.x + threadIdx.x;   // 786432
  int ks    = idx >> 14;
  int rem   = idx & 16383;
  int ntile = rem >> 9;
  int li    = rem & 511;
  int lane  = li >> 4;
  int elem  = li & 15;
  int n    = ntile * 16 + (lane & 15);
  int half = lane >> 4;
  int K    = ks * 32 + half * 16 + elem;
  int k    = K >> 9;          // which tap (0..2)
  int ci   = K & 511;         // input channel
  bws[idx] = f2bf(w[((size_t)n * C_ + ci) * 3 + k]);
}

// ---------- conv-as-GEMM: y[m][n] = relu(sum_kd A[m][kd]*B[kd][n] + bias[n]) ----------
// Double-buffered LDS fed by CDNA5 async global->LDS loads (ASYNCcnt), 8 WMMA/wave/K-step.
__launch_bounds__(256)
__global__ void conv_gemm_kernel(const __bf16* __restrict__ xpad,
                                 const __bf16* __restrict__ bws,
                                 const float* __restrict__ bias,
                                 float* __restrict__ y)
{
  __shared__ alignas(16) __bf16 As[2][128 * 32];   // 2 x 8KB
  __shared__ alignas(16) __bf16 Bs[2][8 * 512];    // 2 x 8KB

  const int tid  = threadIdx.x;
  const int wave = tid >> 5;
  const int lane = tid & 31;
  const int m0   = blockIdx.x * 128;
  const int b    = m0 >> 10;
  const int t0   = m0 & 1023;
  const int nt0  = blockIdx.y * 8;

  const __bf16* Abase = xpad + ((size_t)b * TPAD + t0) * C_;  // A row m=t starts at t*C
  const int arow = tid >> 1;
  const int acol = (tid & 1) * 16;

  // LDS byte offsets for this thread's async-load destinations (low 32 bits of
  // a flat LDS pointer are the LDS offset per the aperture mapping).
  const uint32_t aDst0 = (uint32_t)(uintptr_t)&As[0][arow * 32 + acol];
  const uint32_t bDst0 = (uint32_t)(uintptr_t)&Bs[0][tid * 16];
  const uint32_t aBufStride = (uint32_t)((uintptr_t)&As[1][0] - (uintptr_t)&As[0][0]);
  const uint32_t bBufStride = (uint32_t)((uintptr_t)&Bs[1][0] - (uintptr_t)&Bs[0][0]);

  v8f acc[8];
#pragma unroll
  for (int i = 0; i < 8; ++i) acc[i] = v8f{};

  const int frow = wave * 16 + (lane & 15);
  const int half = lane >> 4;

  // Issue one K-step tile (4 x 16B per thread = 16KB per workgroup) asynchronously.
  auto issue_tile = [&](int buf, int ks) {
    const __bf16* aSrc = Abase + (size_t)arow * C_ + ks * 32 + acol;
    const __bf16* bSrc = bws + (((size_t)(ks * 32 + nt0)) << 9) + tid * 16;
    const uint32_t ad = aDst0 + (uint32_t)buf * aBufStride;
    const uint32_t bd = bDst0 + (uint32_t)buf * bBufStride;
    asm volatile("global_load_async_to_lds_b128 %0, %1, off"
                 :: "v"(ad), "v"(aSrc) : "memory");
    asm volatile("global_load_async_to_lds_b128 %0, %1, off offset:16"
                 :: "v"(ad), "v"(aSrc) : "memory");
    asm volatile("global_load_async_to_lds_b128 %0, %1, off"
                 :: "v"(bd), "v"(bSrc) : "memory");
    asm volatile("global_load_async_to_lds_b128 %0, %1, off offset:16"
                 :: "v"(bd), "v"(bSrc) : "memory");
    if (ks + 1 < KSTEPS) {                       // warm L2 one more step ahead
      __builtin_prefetch(aSrc + 32, 0, 1);
      __builtin_prefetch(bSrc + 32 * 512, 0, 1);
    }
  };

  issue_tile(0, 0);

  for (int ks = 0; ks < KSTEPS; ++ks) {
    const int cur = ks & 1;
    if (ks + 1 < KSTEPS) {
      issue_tile(cur ^ 1, ks + 1);               // copy next tile while we compute
      // async loads complete in order: <=4 outstanding means current tile landed
      asm volatile("s_wait_asynccnt 0x4" ::: "memory");
    } else {
      asm volatile("s_wait_asynccnt 0x0" ::: "memory");
    }
    __syncthreads();                             // all waves' tile data visible

    const __bf16* AsCur = &As[cur][0];
    const __bf16* BsCur = &Bs[cur][0];

    // A fragment (contiguous k = 16*half + s, matching B): two 16B LDS reads per lane
    v8bf a0 = *(const v8bf*)&AsCur[frow * 32 + half * 16];
    v8bf a1 = *(const v8bf*)&AsCur[frow * 32 + half * 16 + 8];
    v16bf a = __builtin_shufflevector(a0, a1, 0,1,2,3,4,5,6,7,8,9,10,11,12,13,14,15);
#pragma unroll
    for (int nt = 0; nt < 8; ++nt) {
      v8bf b0 = *(const v8bf*)&BsCur[nt * 512 + lane * 16];
      v8bf b1 = *(const v8bf*)&BsCur[nt * 512 + lane * 16 + 8];
      v16bf bm = __builtin_shufflevector(b0, b1, 0,1,2,3,4,5,6,7,8,9,10,11,12,13,14,15);
      acc[nt] = __builtin_amdgcn_wmma_f32_16x16x32_bf16(false, a, false, bm,
                                                        (short)0, acc[nt], false, false);
    }
    __syncthreads();                             // reads done before buffer is rewritten
  }

  // epilogue: bias + relu, C/D layout: lane&15 -> N, VGPR r -> M=r + (lane>>4)*8
  const int nlo = lane & 15;
#pragma unroll
  for (int nt = 0; nt < 8; ++nt) {
    const int col = (nt0 + nt) * 16 + nlo;
    const float bv = bias[col];
#pragma unroll
    for (int r = 0; r < 8; ++r) {
      const int gm = m0 + wave * 16 + half * 8 + r;
      float v = acc[nt][r] + bv;
      y[(size_t)gm * C_ + col] = v > 0.f ? v : 0.f;
    }
  }
}

// ---------- LayerNorm(512) -> bf16 into padded buffer (feeds conv2) ----------
__global__ void ln_to_pad_kernel(const float* __restrict__ y,
                                 const float* __restrict__ g,
                                 const float* __restrict__ beta,
                                 __bf16* __restrict__ xpad2)
{
  __shared__ float red[256];
  const int row = blockIdx.x;           // b*T + t
  const int b = row >> 10, t = row & 1023;
  const int tid = threadIdx.x;
  const float* yr = y + (size_t)row * C_;
  float v0 = yr[tid], v1 = yr[tid + 256];
  red[tid] = v0 + v1; __syncthreads();
  for (int o = 128; o > 0; o >>= 1) { if (tid < o) red[tid] += red[tid + o]; __syncthreads(); }
  const float mu = red[0] * (1.f / 512.f);
  __syncthreads();
  const float d0 = v0 - mu, d1 = v1 - mu;
  red[tid] = d0 * d0 + d1 * d1; __syncthreads();
  for (int o = 128; o > 0; o >>= 1) { if (tid < o) red[tid] += red[tid + o]; __syncthreads(); }
  const float rstd = rsqrtf(red[0] * (1.f / 512.f) + 1e-5f);
  __bf16* out = xpad2 + ((size_t)b * TPAD + (t + 1)) * C_;
  out[tid]       = f2bf(d0 * rstd * g[tid]       + beta[tid]);
  out[tid + 256] = f2bf(d1 * rstd * g[tid + 256] + beta[tid + 256]);
}

// ---------- LayerNorm + Linear(512->1) + mask -> predicted_durations ----------
__global__ void ln_linear_kernel(const float* __restrict__ y,
                                 const float* __restrict__ g,
                                 const float* __restrict__ beta,
                                 const float* __restrict__ lin_w,
                                 const float* __restrict__ lin_b,
                                 const int* __restrict__ tlen,
                                 float* __restrict__ pred)
{
  __shared__ float red[256];
  const int row = blockIdx.x;
  const int b = row >> 10, t = row & 1023;
  const int tid = threadIdx.x;
  const float* yr = y + (size_t)row * C_;
  float v0 = yr[tid], v1 = yr[tid + 256];
  red[tid] = v0 + v1; __syncthreads();
  for (int o = 128; o > 0; o >>= 1) { if (tid < o) red[tid] += red[tid + o]; __syncthreads(); }
  const float mu = red[0] * (1.f / 512.f);
  __syncthreads();
  const float d0 = v0 - mu, d1 = v1 - mu;
  red[tid] = d0 * d0 + d1 * d1; __syncthreads();
  for (int o = 128; o > 0; o >>= 1) { if (tid < o) red[tid] += red[tid + o]; __syncthreads(); }
  const float rstd = rsqrtf(red[0] * (1.f / 512.f) + 1e-5f);
  __syncthreads();
  const float h0 = d0 * rstd * g[tid]       + beta[tid];
  const float h1 = d1 * rstd * g[tid + 256] + beta[tid + 256];
  red[tid] = h0 * lin_w[tid] + h1 * lin_w[tid + 256]; __syncthreads();
  for (int o = 128; o > 0; o >>= 1) { if (tid < o) red[tid] += red[tid + o]; __syncthreads(); }
  if (tid == 0) pred[row] = (t < tlen[b]) ? (red[0] + lin_b[0]) : 0.f;
}

// ---------- per-batch duration prefix sum + mel length ----------
__global__ void scan_kernel(const int* __restrict__ dur, const int* __restrict__ tlen,
                            int* __restrict__ cum, float* __restrict__ mel)
{
  __shared__ int sb[1024];
  const int b = blockIdx.x, t = threadIdx.x;
  const int len = tlen[b];
  const int valid = (t < len) ? 1 : 0;
  int d = valid ? dur[b * T_ + t] : 0;
  sb[t] = d; __syncthreads();
  for (int o = 512; o > 0; o >>= 1) { if (t < o) sb[t] += sb[t + o]; __syncthreads(); }
  const int total = sb[0]; __syncthreads();
  if (total == 0) d = valid;                      // torch: all-zero durations -> ones
  sb[t] = d; __syncthreads();
  for (int o = 1; o < 1024; o <<= 1) {            // Hillis-Steele inclusive scan
    int self = sb[t];
    int othr = (t >= o) ? sb[t - o] : 0;
    __syncthreads();
    sb[t] = self + othr;
    __syncthreads();
  }
  cum[b * T_ + t] = sb[t];
  if (t == 1023) mel[b] = (float)sb[1023];
}

// ---------- regulator gather: out[b,f,:] = batch[b, searchsorted(cum,f), :] ----------
__global__ void gather_kernel(const float* __restrict__ batch, const int* __restrict__ cum,
                              float* __restrict__ out)
{
  const int f = blockIdx.x & (LMAX - 1);
  const int b = blockIdx.x >> 12;
  const int* c = cum + b * T_;
  const int total = c[T_ - 1];
  float4* dst = (float4*)out + ((size_t)b * LMAX + f) * 128 + threadIdx.x;
  if (f >= total) { *dst = make_float4(0.f, 0.f, 0.f, 0.f); return; }
  int lo = 0, hi = T_;                             // first idx with cum[idx] > f
  while (lo < hi) { int mid = (lo + hi) >> 1; if (c[mid] <= f) lo = mid + 1; else hi = mid; }
  int idx = lo > T_ - 1 ? T_ - 1 : lo;
  const float4* src = (const float4*)batch + ((size_t)b * T_ + idx) * 128 + threadIdx.x;
  *dst = *src;
}

extern "C" void kernel_launch(void* const* d_in, const int* in_sizes, int n_in,
                              void* d_out, int out_size, void* d_ws, size_t ws_size,
                              hipStream_t stream)
{
  const float* batch = (const float*)d_in[0];
  const int*   tlen  = (const int*)d_in[1];
  // d_in[2] = mask (recomputed from token_lengths)
  const int*   durs  = (const int*)d_in[3];
  const float* w1  = (const float*)d_in[4];
  const float* b1  = (const float*)d_in[5];
  const float* g1  = (const float*)d_in[6];
  const float* be1 = (const float*)d_in[7];
  const float* w2  = (const float*)d_in[8];
  const float* b2  = (const float*)d_in[9];
  const float* g2  = (const float*)d_in[10];
  const float* be2 = (const float*)d_in[11];
  const float* lw  = (const float*)d_in[12];
  const float* lb  = (const float*)d_in[13];

  char* ws = (char*)d_ws;
  __bf16* xpad1 = (__bf16*)(ws);
  __bf16* xpad2 = (__bf16*)(ws + XPAD_ELEMS * 2);
  __bf16* bws1  = (__bf16*)(ws + XPAD_ELEMS * 4);
  __bf16* bws2  = (__bf16*)(ws + XPAD_ELEMS * 4 + BWS_ELEMS * 2);
  float*  yy    = (float*)(ws + XPAD_ELEMS * 4 + BWS_ELEMS * 4);
  int*    cum   = (int*)(ws + XPAD_ELEMS * 4 + BWS_ELEMS * 4 + Y_ELEMS * 4);

  float* out_pad  = (float*)d_out;
  float* out_mel  = out_pad + (size_t)B_ * LMAX * C_;
  float* out_pred = out_mel + B_;

  pad_convert_kernel<<<(unsigned)((XPAD_ELEMS / 4 + 255) / 256), 256, 0, stream>>>(batch, xpad1);
  zero_pad_kernel<<<(B_ * 2 * C_) / 256, 256, 0, stream>>>(xpad2);
  weight_prep_kernel<<<(unsigned)(BWS_ELEMS / 256), 256, 0, stream>>>(w1, bws1);
  weight_prep_kernel<<<(unsigned)(BWS_ELEMS / 256), 256, 0, stream>>>(w2, bws2);

  dim3 ggrid(B_ * T_ / 128, C_ / 128);
  conv_gemm_kernel<<<ggrid, 256, 0, stream>>>(xpad1, bws1, b1, yy);
  ln_to_pad_kernel<<<B_ * T_, 256, 0, stream>>>(yy, g1, be1, xpad2);
  conv_gemm_kernel<<<ggrid, 256, 0, stream>>>(xpad2, bws2, b2, yy);
  ln_linear_kernel<<<B_ * T_, 256, 0, stream>>>(yy, g2, be2, lw, lb, tlen, out_pred);

  scan_kernel<<<B_, 1024, 0, stream>>>(durs, tlen, cum, out_mel);
  gather_kernel<<<B_ * LMAX, 128, 0, stream>>>(batch, cum, out_pad);
}
